// Block_38345468019184
// MI455X (gfx1250) — matrix operand order repaired and checked by reading, not compiled, not run on hardware
//
#include <hip/hip_runtime.h>
#include <hip/hip_bf16.h>
#include <math.h>

// ---------------------------------------------------------------------------
// Transformer block forward for MI455X (gfx1250, wave32, WMMA).
// - All GEMMs + attention matmuls: v_wmma_f32_16x16x32_bf16 (f32 accum)
// - GEMM tile staging: TDM tensor_load_to_lds (TENSORcnt), double-buffered
// - Attention staging: GLOBAL_LOAD_ASYNC_TO_LDS_B128 (ASYNCcnt)
// - Softmax reductions: v_permlane16_b32 XOR patterns (VALU, no LDS traffic)
// ---------------------------------------------------------------------------

typedef __attribute__((ext_vector_type(16))) __bf16 v16bf;
typedef __attribute__((ext_vector_type(8)))  float  v8f;
typedef __attribute__((ext_vector_type(4)))  unsigned int u32x4;
typedef __attribute__((ext_vector_type(8)))  int    i32x8;
typedef __attribute__((ext_vector_type(4)))  int    i32x4;

#define B_DIM 4
#define T_DIM 2048
#define H_DIM 16
#define D_DIM 64
#define C_DIM 1024
#define FF_DIM 4096
#define M_DIM (B_DIM * T_DIM)   // 8192 rows

union FragBF { v16bf v; __bf16 h[16]; };

static __device__ __forceinline__ v8f v8f_zero() {
  v8f z = {0.f, 0.f, 0.f, 0.f, 0.f, 0.f, 0.f, 0.f};
  return z;
}

static __device__ __forceinline__ v8f wmma_bf16(const v16bf a, const v16bf b, const v8f c) {
  return __builtin_amdgcn_wmma_f32_16x16x32_bf16(false, a, false, b, (short)0, c, false, false);
}

// LDS byte offset of a __shared__ pointer (generic LDS addr low 32 bits == LDS offset)
static __device__ __forceinline__ unsigned lds_off(const void* p) {
  return (unsigned)(uintptr_t)p;
}

// async 16B copy: global -> LDS, tracked by ASYNCcnt
static __device__ __forceinline__ void async_cp16(unsigned lds_byte, const void* gptr) {
  asm volatile("global_load_async_to_lds_b128 %0, %1, off"
               :: "v"(lds_byte), "v"(gptr) : "memory");
}

// v_permlane16_b32 lane-xor within each 16-lane row (pure VALU shuffle)
static __device__ __forceinline__ float permxor16(float x, int s0, int s1) {
  return __int_as_float(__builtin_amdgcn_permlane16(
      __float_as_int(x), __float_as_int(x), s0, s1, false, false));
}
// selector nibble tables for sel[i] = i ^ d within a 16-lane row
#define XSEL1_LO 0x67452301
#define XSEL1_HI 0xEFCDAB89
#define XSEL2_LO 0x54761032
#define XSEL2_HI 0xDCFE98BA
#define XSEL4_LO 0x32107654
#define XSEL4_HI 0xBA98FEDC
#define XSEL8_LO 0xFEDCBA98
#define XSEL8_HI 0x76543210

static __device__ __forceinline__ float rowmax16(float t) {
  t = fmaxf(t, permxor16(t, XSEL1_LO, XSEL1_HI));
  t = fmaxf(t, permxor16(t, XSEL2_LO, XSEL2_HI));
  t = fmaxf(t, permxor16(t, XSEL4_LO, XSEL4_HI));
  t = fmaxf(t, permxor16(t, XSEL8_LO, XSEL8_HI));
  return t;
}
static __device__ __forceinline__ float rowsum16(float t) {
  t += permxor16(t, XSEL1_LO, XSEL1_HI);
  t += permxor16(t, XSEL2_LO, XSEL2_HI);
  t += permxor16(t, XSEL4_LO, XSEL4_HI);
  t += permxor16(t, XSEL8_LO, XSEL8_HI);
  return t;
}

// ---------------------------------------------------------------------------
// TDM descriptor helpers (D# groups, ISA cdna5 ch.8)
// ---------------------------------------------------------------------------
static __device__ __forceinline__ u32x4 tdm_g0(unsigned lds_byte, const void* gptr) {
  unsigned long long ga = (unsigned long long)(uintptr_t)gptr;
  u32x4 g;
  g[0] = 1u;                                   // count=1 (valid), user mode
  g[1] = lds_byte;                             // lds_addr
  g[2] = (unsigned)ga;                         // global_addr[31:0]
  g[3] = (unsigned)((ga >> 32) & 0x01FFFFFFu)  // global_addr[56:32]
         | (2u << 30);                         // type=2 ("image")
  return g;
}
// 2D tile: tile_dim0 x tile_dim1 elements (2B each), row stride = stride0 elems
static __device__ __forceinline__ i32x8 tdm_g1_2d(unsigned td0, unsigned td1,
                                                  unsigned tile0, unsigned tile1,
                                                  unsigned long long stride0) {
  i32x8 g;
  g[0] = (int)(1u << 16);                                           // data_size=1 (2B)
  g[1] = (int)((td0 & 0xFFFFu) << 16);                              // tensor_dim0 lo
  g[2] = (int)(((td0 >> 16) & 0xFFFFu) | ((td1 & 0xFFFFu) << 16));  // td0 hi | td1 lo
  g[3] = (int)(((td1 >> 16) & 0xFFFFu) | ((tile0 & 0xFFFFu) << 16));
  g[4] = (int)(tile1 & 0xFFFFu);                                    // tile_dim1, tile_dim2=0
  g[5] = (int)(unsigned)(stride0 & 0xFFFFFFFFull);                  // dim0 stride lo
  g[6] = (int)(unsigned)((stride0 >> 32) & 0xFFFFull);              // dim0 stride hi
  g[7] = 0;                                                         // dim1 stride (unused, 2D)
  return g;
}

// issue one 2D-tile TDM load (6-arg builtin: g0, g1, g2, g3, g4, cpol)
static __device__ __forceinline__ void tdm_load(unsigned lds_byte, const void* gptr,
                                                const i32x8 g1) {
  const i32x4 z4 = {0, 0, 0, 0};
  const i32x8 z8 = {0, 0, 0, 0, 0, 0, 0, 0};
  __builtin_amdgcn_tensor_load_to_lds(tdm_g0(lds_byte, gptr), g1, z4, z4, z8, 0);
}

// ---------------------------------------------------------------------------
// fp32 -> bf16 cast
// ---------------------------------------------------------------------------
__global__ void cast_f32_to_bf16(const float* __restrict__ in,
                                 __bf16* __restrict__ out, size_t n) {
  size_t i = (size_t)blockIdx.x * blockDim.x + threadIdx.x;
  size_t stride = (size_t)gridDim.x * blockDim.x;
  for (; i < n; i += stride) out[i] = (__bf16)in[i];
}

// ---------------------------------------------------------------------------
// LayerNorm over C=1024, one block (256 thr) per row, bf16 output
// ---------------------------------------------------------------------------
__global__ __launch_bounds__(256) void layernorm_bf16(
    const float* __restrict__ x, const float* __restrict__ g,
    const float* __restrict__ b, __bf16* __restrict__ out) {
  __shared__ float red[256];
  const int tid = threadIdx.x;
  const size_t base = (size_t)blockIdx.x * C_DIM;

  float v[4];
  float s = 0.f;
#pragma unroll
  for (int i = 0; i < 4; ++i) {
    v[i] = x[base + tid + i * 256];
    s += v[i];
  }
  red[tid] = s;
  __syncthreads();
  for (int off = 128; off > 0; off >>= 1) {
    if (tid < off) red[tid] += red[tid + off];
    __syncthreads();
  }
  const float mean = red[0] * (1.0f / C_DIM);
  __syncthreads();

  float vs = 0.f;
#pragma unroll
  for (int i = 0; i < 4; ++i) {
    float d = v[i] - mean;
    vs += d * d;
  }
  red[tid] = vs;
  __syncthreads();
  for (int off = 128; off > 0; off >>= 1) {
    if (tid < off) red[tid] += red[tid + off];
    __syncthreads();
  }
  const float rstd = rsqrtf(red[0] * (1.0f / C_DIM) + 1e-5f);
  __syncthreads();

#pragma unroll
  for (int i = 0; i < 4; ++i) {
    int c = tid + i * 256;
    out[base + c] = (__bf16)((v[i] - mean) * rstd * g[c] + b[c]);
  }
}

// ---------------------------------------------------------------------------
// Tiled WMMA GEMM:  out = epilogue(A[M,K] @ B[K,N] + bias [+ res])
// Block tile 128x128, BK=32. 8 waves (2x4), wave tile 64x32 (4x2 WMMA accs).
// Tile staging via TDM: wave 0 issues 2 tensor_load_to_lds per stage,
// double-buffered, synchronized with s_wait_tensorcnt + barrier.
// ---------------------------------------------------------------------------
template <bool GELU_, bool RES_, bool OUTBF_>
__global__ __launch_bounds__(256) void gemm_bf16_wmma(
    const __bf16* __restrict__ A, const __bf16* __restrict__ Bm,
    const float* __restrict__ bias, const float* __restrict__ res,
    float* __restrict__ outF, __bf16* __restrict__ outB,
    int M, int N, int K) {
  __shared__ __bf16 As[2][128 * 32];  // [buf][m][k]  2x8KB
  __shared__ __bf16 Bs[2][32 * 128];  // [buf][k][n]  2x8KB

  const int tid  = threadIdx.x;
  const int lane = tid & 31;
  const int lm   = lane & 15;
  const int half = lane >> 4;
  const int wave = tid >> 5;
  const int wm   = wave >> 2;  // 0..1
  const int wn   = wave & 3;   // 0..3
  const int m0   = blockIdx.y * 128;
  const int n0   = blockIdx.x * 128;

  const unsigned asBase[2] = {lds_off(&As[0][0]), lds_off(&As[1][0])};
  const unsigned bsBase[2] = {lds_off(&Bs[0][0]), lds_off(&Bs[1][0])};

  // stage-invariant descriptor group-1 words
  const i32x8 g1A = tdm_g1_2d(/*td0=*/32,  /*td1=*/128, /*tile0=*/32,  /*tile1=*/128,
                              (unsigned long long)K);
  const i32x8 g1B = tdm_g1_2d(/*td0=*/128, /*td1=*/32,  /*tile0=*/128, /*tile1=*/32,
                              (unsigned long long)N);

  v8f acc[4][2];
#pragma unroll
  for (int i = 0; i < 4; ++i)
#pragma unroll
    for (int j = 0; j < 2; ++j) acc[i][j] = v8f_zero();

  const int nIt = K >> 5;  // K/32

  // prime stage 0
  if (wave == 0) {
    tdm_load(asBase[0], A + (size_t)m0 * K, g1A);
    tdm_load(bsBase[0], Bm + n0, g1B);
  }

  for (int it = 0; it < nIt; ++it) {
    const int cur = it & 1;
    if (it + 1 < nIt) {
      if (wave == 0) {
        const int k1 = (it + 1) * 32;
        tdm_load(asBase[cur ^ 1], A + (size_t)m0 * K + k1, g1A);
        tdm_load(bsBase[cur ^ 1], Bm + (size_t)k1 * N + n0, g1B);
      }
      __builtin_amdgcn_s_wait_tensorcnt(2);  // TDM in-order: stage `it` landed
    } else {
      __builtin_amdgcn_s_wait_tensorcnt(0);
    }
    __syncthreads();  // stage `it` visible to all waves

    const __bf16* __restrict__ as = &As[cur][0];
    const __bf16* __restrict__ bs = &Bs[cur][0];

    // A fragments (16-bit A 16x32: lane m = lane%16, half picks K sub-block)
    FragBF af[4];
#pragma unroll
    for (int mf = 0; mf < 4; ++mf) {
      int row = wm * 64 + mf * 16 + lm;
#pragma unroll
      for (int g = 0; g < 8; ++g) {
        int kk = 2 * g + 16 * half;
        af[mf].h[2 * g]     = as[row * 32 + kk];
        af[mf].h[2 * g + 1] = as[row * 32 + kk + 1];
      }
    }
    // B fragments (16-bit B 32x16: lane n = lane%16, half picks K 0..15/16..31)
    FragBF bfr[2];
#pragma unroll
    for (int nf = 0; nf < 2; ++nf) {
      int col = wn * 32 + nf * 16 + lm;
#pragma unroll
      for (int g = 0; g < 8; ++g) {
        int kk = 2 * g + 16 * half;
        bfr[nf].h[2 * g]     = bs[kk * 128 + col];
        bfr[nf].h[2 * g + 1] = bs[(kk + 1) * 128 + col];
      }
    }
#pragma unroll
    for (int mf = 0; mf < 4; ++mf)
#pragma unroll
      for (int nf = 0; nf < 2; ++nf)
        acc[mf][nf] = wmma_bf16(af[mf].v, bfr[nf].v, acc[mf][nf]);
    __syncthreads();  // done reading buf `cur` before it is refilled
  }

  // epilogue (C/D layout: VGPR j, lane -> row m=j+8*half, col n=lane%16)
#pragma unroll
  for (int mf = 0; mf < 4; ++mf) {
#pragma unroll
    for (int nf = 0; nf < 2; ++nf) {
      int cidx = n0 + wn * 32 + nf * 16 + lm;
#pragma unroll
      for (int j = 0; j < 8; ++j) {
        int r = m0 + wm * 64 + mf * 16 + j + 8 * half;
        float val = acc[mf][nf][j] + bias[cidx];
        if (GELU_) val = 0.5f * val * (1.0f + erff(val * 0.70710678118654752f));
        if (RES_) val += res[(size_t)r * N + cidx];
        if (OUTBF_) outB[(size_t)r * N + cidx] = (__bf16)val;
        else        outF[(size_t)r * N + cidx] = val;
      }
    }
  }
}

// ---------------------------------------------------------------------------
// Flash attention (causal). One block = 128 query rows of one (b,h).
// 8 waves x 16 query rows; KV streamed in 32-row tiles through LDS (async).
// S = Q@K^T and O += P@V both on WMMA bf16.
// ---------------------------------------------------------------------------
__global__ __launch_bounds__(256) void flash_attn_wmma(
    const __bf16* __restrict__ qkv,  // [B*T, 3*C], q|k|v packed
    __bf16* __restrict__ y) {        // [B*T, C]
  __shared__ __bf16 Qs[128 * 64];    // 16 KB
  __shared__ __bf16 Ks[32 * 64];     //  4 KB
  __shared__ __bf16 Vs[32 * 64];     //  4 KB
  __shared__ __bf16 Ps[8][16 * 32];  //  8 KB, wave-private P scratch

  const int tid  = threadIdx.x;
  const int lane = tid & 31;
  const int lm   = lane & 15;
  const int half = lane >> 4;
  const int wave = tid >> 5;

  const int bh = blockIdx.y;
  const int b  = bh / H_DIM;
  const int h  = bh % H_DIM;
  const int q0 = blockIdx.x * 128;
  const size_t rowstride = (size_t)3 * C_DIM;
  const __bf16* basep = qkv + (size_t)b * T_DIM * rowstride + h * D_DIM;

  const unsigned qsBase = lds_off(&Qs[0]);
  const unsigned ksBase = lds_off(&Ks[0]);
  const unsigned vsBase = lds_off(&Vs[0]);

  // ---- async load Q tile (128x64 bf16 = 1024 x 16B chunks, 4/thread) ----
#pragma unroll
  for (int i = 0; i < 4; ++i) {
    int c = i * 256 + tid;
    int row = c >> 3, dc = (c & 7) * 8;
    async_cp16(qsBase + (unsigned)(row * 64 + dc) * 2,
               basep + (size_t)(q0 + row) * rowstride + dc);
  }
  asm volatile("s_wait_asynccnt 0x0" ::: "memory");
  __syncthreads();

  // ---- Q A-fragments: 16x64 per wave -> two K=32 fragments ----
  FragBF qa[2];
#pragma unroll
  for (int ks = 0; ks < 2; ++ks) {
    int row = wave * 16 + lm;
#pragma unroll
    for (int g = 0; g < 8; ++g) {
      int kk = 2 * g + 16 * half;
      qa[ks].h[2 * g]     = Qs[row * 64 + ks * 32 + kk];
      qa[ks].h[2 * g + 1] = Qs[row * 64 + ks * 32 + kk + 1];
    }
  }

  float mi[8], li[8];
  v8f oacc[4];
#pragma unroll
  for (int j = 0; j < 8; ++j) { mi[j] = -1e30f; li[j] = 0.f; }
#pragma unroll
  for (int d = 0; d < 4; ++d) oacc[d] = v8f_zero();

  const float scale = 0.125f;  // 1/sqrt(64)
  const int q_hi = q0 + 127;   // block-uniform causal bound

  for (int kv0 = 0; kv0 <= q_hi; kv0 += 32) {
    __syncthreads();  // everyone done with previous K/V tiles
    // ---- async load K,V tiles (32x64 each = 256 x 16B chunks, 1/thread) ----
    {
      int row = tid >> 3, dc = (tid & 7) * 8;
      const __bf16* kvrow = basep + (size_t)(kv0 + row) * rowstride + dc;
      async_cp16(ksBase + (unsigned)(row * 64 + dc) * 2, kvrow + C_DIM);
      async_cp16(vsBase + (unsigned)(row * 64 + dc) * 2, kvrow + 2 * C_DIM);
    }
    asm volatile("s_wait_asynccnt 0x0" ::: "memory");
    __syncthreads();

    // ---- S = Q @ K^T : two 16x16 tiles (keys kv0..+15, kv0+16..+31) ----
    v8f sacc[2];
    sacc[0] = v8f_zero();
    sacc[1] = v8f_zero();
#pragma unroll
    for (int nf = 0; nf < 2; ++nf) {
#pragma unroll
      for (int ks = 0; ks < 2; ++ks) {
        FragBF kb;  // B matrix: (d 32) x (16 keys); lane n = key
        int key = nf * 16 + lm;
#pragma unroll
        for (int g = 0; g < 8; ++g) {
          int dd = ks * 32 + 2 * g + 16 * half;
          kb.h[2 * g]     = Ks[key * 64 + dd];
          kb.h[2 * g + 1] = Ks[key * 64 + dd + 1];
        }
        sacc[nf] = wmma_bf16(qa[ks].v, kb.v, sacc[nf]);
      }
    }

    // ---- online softmax (rows j+8*half, cols = lane%16 per tile) ----
#pragma unroll
    for (int j = 0; j < 8; ++j) {
      int grow = q0 + wave * 16 + j + 8 * half;
      int key0 = kv0 + lm;
      int key1 = kv0 + 16 + lm;
      float s0 = sacc[0][j] * scale;
      float s1 = sacc[1][j] * scale;
      bool msk0 = key0 > grow, msk1 = key1 > grow;
      if (msk0) s0 = -1e30f;
      if (msk1) s1 = -1e30f;

      float t = rowmax16(fmaxf(s0, s1));       // v_permlane16 reduction
      float mn = fmaxf(mi[j], t);
      float alpha = __expf(mi[j] - mn);
      float p0 = msk0 ? 0.f : __expf(s0 - mn);
      float p1 = msk1 ? 0.f : __expf(s1 - mn);
      float rs = rowsum16(p0 + p1);            // v_permlane16 reduction
      li[j] = li[j] * alpha + rs;
      mi[j] = mn;
#pragma unroll
      for (int df = 0; df < 4; ++df) oacc[df][j] *= alpha;

      int mloc = j + 8 * half;
      Ps[wave][mloc * 32 + lm]      = (__bf16)p0;
      Ps[wave][mloc * 32 + 16 + lm] = (__bf16)p1;
    }

    // wave-private LDS scratch: DS ops are in-order per wave; drain writes
    asm volatile("s_wait_dscnt 0x0" ::: "memory");

    // ---- O += P @ V ----
    FragBF pa;  // A matrix 16x32
#pragma unroll
    for (int g = 0; g < 8; ++g) {
      int kk = 2 * g + 16 * half;
      pa.h[2 * g]     = Ps[wave][lm * 32 + kk];
      pa.h[2 * g + 1] = Ps[wave][lm * 32 + kk + 1];
    }
#pragma unroll
    for (int df = 0; df < 4; ++df) {
      FragBF vb;  // B matrix: (key 32) x (16 dims)
      int col = df * 16 + lm;
#pragma unroll
      for (int g = 0; g < 8; ++g) {
        int kk = 2 * g + 16 * half;
        vb.h[2 * g]     = Vs[kk * 64 + col];
        vb.h[2 * g + 1] = Vs[(kk + 1) * 64 + col];
      }
      oacc[df] = wmma_bf16(pa.v, vb.v, oacc[df]);
    }
  }

  // ---- epilogue: normalize rows, write y[b*T+row, h*64 + d] as bf16 ----
#pragma unroll
  for (int df = 0; df < 4; ++df) {
#pragma unroll
    for (int j = 0; j < 8; ++j) {
      int grow = q0 + wave * 16 + j + 8 * half;
      int col = h * D_DIM + df * 16 + lm;
      float o = oacc[df][j] / li[j];
      y[((size_t)b * T_DIM + grow) * C_DIM + col] = (__bf16)o;
    }
  }
}

// ---------------------------------------------------------------------------
// Host orchestration
// ---------------------------------------------------------------------------
extern "C" void kernel_launch(void* const* d_in, const int* in_sizes, int n_in,
                              void* d_out, int out_size, void* d_ws,
                              size_t ws_size, hipStream_t stream) {
  (void)in_sizes; (void)n_in; (void)out_size; (void)ws_size;

  const float* x     = (const float*)d_in[0];
  const float* ln1w  = (const float*)d_in[1];
  const float* ln1b  = (const float*)d_in[2];
  const float* w_qkv = (const float*)d_in[3];
  const float* b_qkv = (const float*)d_in[4];
  const float* w_o   = (const float*)d_in[5];
  const float* b_o   = (const float*)d_in[6];
  const float* ln2w  = (const float*)d_in[7];
  const float* ln2b  = (const float*)d_in[8];
  const float* w_fc  = (const float*)d_in[9];
  const float* b_fc  = (const float*)d_in[10];
  const float* w_out = (const float*)d_in[11];
  const float* b_out = (const float*)d_in[12];
  float* out = (float*)d_out;

  // workspace carve-up (256B aligned)
  char* ws = (char*)d_ws;
  size_t off = 0;
  auto take = [&](size_t bytes) -> char* {
    char* p = ws + off;
    off = (off + bytes + 255) & ~(size_t)255;
    return p;
  };
  __bf16* wqkv_bf = (__bf16*)take((size_t)C_DIM * 3 * C_DIM * 2);
  __bf16* wo_bf   = (__bf16*)take((size_t)C_DIM * C_DIM * 2);
  __bf16* wfc_bf  = (__bf16*)take((size_t)C_DIM * FF_DIM * 2);
  __bf16* wout_bf = (__bf16*)take((size_t)FF_DIM * C_DIM * 2);
  __bf16* xln     = (__bf16*)take((size_t)M_DIM * C_DIM * 2);   // reused ln1/ln2
  __bf16* qkv_bf  = (__bf16*)take((size_t)M_DIM * 3 * C_DIM * 2);
  __bf16* y_bf    = (__bf16*)take((size_t)M_DIM * C_DIM * 2);
  float*  x2      = (float*)take((size_t)M_DIM * C_DIM * 4);
  __bf16* h_bf    = (__bf16*)take((size_t)M_DIM * FF_DIM * 2);

  // 1) cast weights fp32 -> bf16
  cast_f32_to_bf16<<<512, 256, 0, stream>>>(w_qkv, wqkv_bf, (size_t)C_DIM * 3 * C_DIM);
  cast_f32_to_bf16<<<512, 256, 0, stream>>>(w_o,   wo_bf,   (size_t)C_DIM * C_DIM);
  cast_f32_to_bf16<<<512, 256, 0, stream>>>(w_fc,  wfc_bf,  (size_t)C_DIM * FF_DIM);
  cast_f32_to_bf16<<<512, 256, 0, stream>>>(w_out, wout_bf, (size_t)FF_DIM * C_DIM);

  // 2) LN1(x) -> xln (bf16)
  layernorm_bf16<<<M_DIM, 256, 0, stream>>>(x, ln1w, ln1b, xln);

  // 3) qkv = xln @ w_qkv + b_qkv  (bf16 out)
  gemm_bf16_wmma<false, false, true><<<dim3(3 * C_DIM / 128, M_DIM / 128), 256, 0, stream>>>(
      xln, wqkv_bf, b_qkv, nullptr, nullptr, qkv_bf, M_DIM, 3 * C_DIM, C_DIM);

  // 4) flash attention -> y_bf
  flash_attn_wmma<<<dim3(T_DIM / 128, B_DIM * H_DIM), 256, 0, stream>>>(qkv_bf, y_bf);

  // 5) x2 = x + y @ w_o + b_o  (f32 out)
  gemm_bf16_wmma<false, true, false><<<dim3(C_DIM / 128, M_DIM / 128), 256, 0, stream>>>(
      y_bf, wo_bf, b_o, x, x2, nullptr, M_DIM, C_DIM, C_DIM);

  // 6) LN2(x2) -> xln (bf16)
  layernorm_bf16<<<M_DIM, 256, 0, stream>>>(x2, ln2w, ln2b, xln);

  // 7) h = gelu(xln @ w_fc + b_fc)  (bf16 out)
  gemm_bf16_wmma<true, false, true><<<dim3(FF_DIM / 128, M_DIM / 128), 256, 0, stream>>>(
      xln, wfc_bf, b_fc, nullptr, nullptr, h_bf, M_DIM, FF_DIM, C_DIM);

  // 8) out = x2 + h @ w_out + b_out  (f32 out)
  gemm_bf16_wmma<false, true, false><<<dim3(C_DIM / 128, M_DIM / 128), 256, 0, stream>>>(
      h_bf, wout_bf, b_out, x2, out, nullptr, M_DIM, C_DIM, FF_DIM);
}
